// Spatial_Attention_layer_33526514712802
// MI455X (gfx1250) — matrix operand order repaired and checked by reading, not compiled, not run on hardware
//
#include <hip/hip_runtime.h>

typedef __attribute__((ext_vector_type(16))) __bf16 v16bf;
typedef __attribute__((ext_vector_type(8)))  __bf16 v8bf;
typedef __attribute__((ext_vector_type(8)))  float  v8f;

#define B_   16
#define N_   2048
#define F_   64
#define LOG2E 1.44269504088896340736f
#define ALPHA 0.2f

// v(t) = exp(sigmoid(leaky_relu(t)))  -- softmax numerator (max-shift free: sigmoid in (0,1))
__device__ __forceinline__ float att_val(float t) {
    float lr  = fmaxf(t, ALPHA * t);                                   // leaky relu (alpha<1)
    float en  = __builtin_amdgcn_exp2f(-lr * LOG2E);                   // exp(-lr)
    float sig = __builtin_amdgcn_rcpf(1.0f + en);                      // sigmoid
    return __builtin_amdgcn_exp2f(sig * LOG2E);                       // exp(sigmoid)
}

// ---------------- Kernel 1: Wh = h@W ; x = Wh@a1 ; y = Wh@a2 ; WhT(bf16) ----------------
__global__ __launch_bounds__(128) void k1_project(const float* __restrict__ h,
                                                  const float* __restrict__ W,
                                                  const float* __restrict__ a,
                                                  __bf16* __restrict__ WhT,
                                                  float* __restrict__ x,
                                                  float* __restrict__ y) {
    __shared__ alignas(16) float Wl[F_ * F_];     // 16 KB
    __shared__ float al[2 * F_];
    __shared__ float hl[128 * 65];                // padded rows: conflict-free dynamic k

    int tid = threadIdx.x;
    for (int q = tid; q < F_ * F_; q += 128) Wl[q] = W[q];
    if (tid < 2 * F_) al[tid] = a[tid];

    int row = blockIdx.x * 128 + tid;             // 0 .. B*N-1
    int b   = row >> 11;
    int i   = row & (N_ - 1);

    const float4* hr4 = (const float4*)(h + (size_t)row * F_);
#pragma unroll
    for (int q = 0; q < 16; ++q) {
        float4 v = hr4[q];
        float* p = hl + tid * 65 + 4 * q;
        p[0] = v.x; p[1] = v.y; p[2] = v.z; p[3] = v.w;
    }
    __syncthreads();

    float4 wh[16];
#pragma unroll
    for (int q = 0; q < 16; ++q) wh[q] = make_float4(0.f, 0.f, 0.f, 0.f);

    for (int k = 0; k < F_; ++k) {
        float hk = hl[tid * 65 + k];
        const float4* wr = (const float4*)(Wl + (k << 6));   // W row k (uniform -> LDS broadcast)
#pragma unroll
        for (int f4 = 0; f4 < 16; ++f4) {
            float4 w = wr[f4];
            wh[f4].x += hk * w.x;  wh[f4].y += hk * w.y;
            wh[f4].z += hk * w.z;  wh[f4].w += hk * w.w;
        }
    }

    float xv = 0.f, yv = 0.f;
#pragma unroll
    for (int f4 = 0; f4 < 16; ++f4) {
        xv += wh[f4].x * al[4*f4+0] + wh[f4].y * al[4*f4+1]
            + wh[f4].z * al[4*f4+2] + wh[f4].w * al[4*f4+3];
        yv += wh[f4].x * al[F_+4*f4+0] + wh[f4].y * al[F_+4*f4+1]
            + wh[f4].z * al[F_+4*f4+2] + wh[f4].w * al[F_+4*f4+3];
    }
    x[row] = xv;
    y[row] = yv;

    __bf16* wt = WhT + (size_t)b * F_ * N_ + i;   // [b][f][i] transposed, bf16
#pragma unroll
    for (int f4 = 0; f4 < 16; ++f4) {
        wt[(size_t)(4*f4+0) * N_] = (__bf16)wh[f4].x;
        wt[(size_t)(4*f4+1) * N_] = (__bf16)wh[f4].y;
        wt[(size_t)(4*f4+2) * N_] = (__bf16)wh[f4].z;
        wt[(size_t)(4*f4+3) * N_] = (__bf16)wh[f4].w;
    }
}

// ---------------- Kernel 2: rD[b][j] = 1 / sum_i v(x_i + y_j) ----------------
__global__ __launch_bounds__(256) void k2_colsum(const float* __restrict__ x,
                                                 const float* __restrict__ y,
                                                 float* __restrict__ rD) {
    __shared__ alignas(16) float xl[N_];          // 8 KB: all x for this batch
    int b = blockIdx.y;
    int tid = threadIdx.x;
    const float4* xb4 = (const float4*)(x + (size_t)b * N_);
    for (int q = tid; q < N_ / 4; q += 256) ((float4*)xl)[q] = xb4[q];
    __syncthreads();

    int j = blockIdx.x * 256 + tid;
    float yj = y[(size_t)b * N_ + j];
    float d = 0.f;
    for (int i4 = 0; i4 < N_ / 4; ++i4) {
        float4 xv = ((const float4*)xl)[i4];
        d += att_val(xv.x + yj);
        d += att_val(xv.y + yj);
        d += att_val(xv.z + yj);
        d += att_val(xv.w + yj);
    }
    rD[(size_t)b * N_ + j] = __builtin_amdgcn_rcpf(d);
}

#define LD8(dst, ptr)                                                           \
    { float4 _a = ((const float4*)(ptr))[0], _b = ((const float4*)(ptr))[1];    \
      dst[0]=_a.x; dst[1]=_a.y; dst[2]=_a.z; dst[3]=_a.w;                       \
      dst[4]=_b.x; dst[5]=_b.y; dst[6]=_b.z; dst[7]=_b.w; }

// ---------------- Kernel 3: out = ELU( A @ Wh ), A generated on the fly ----------------
// Block: 4 waves; wave w owns rows [blk*64 + 16w, +16), all 64 feature cols.
__global__ __launch_bounds__(128) void k3_attn_gemm(const __bf16* __restrict__ WhT,
                                                    const float* __restrict__ x,
                                                    const float* __restrict__ y,
                                                    const float* __restrict__ rD,
                                                    float* __restrict__ out) {
    __shared__ alignas(16) float yl[N_];          // 8 KB
    __shared__ alignas(16) float rl[N_];          // 8 KB
    int b = blockIdx.y;
    int tid = threadIdx.x;
    const float4* y4 = (const float4*)(y  + (size_t)b * N_);
    const float4* r4 = (const float4*)(rD + (size_t)b * N_);
    for (int q = tid; q < N_ / 4; q += 128) {
        ((float4*)yl)[q] = y4[q];
        ((float4*)rl)[q] = r4[q];
    }
    __syncthreads();

    int wave = tid >> 5, lane = tid & 31;
    int mloc = lane & 15, hi = lane >> 4;
    int i0 = blockIdx.x * 64 + wave * 16;         // this wave's m-tile row base
    float xv = x[(size_t)b * N_ + i0 + mloc];     // A lane's row scalar (fixed over K)

    const __bf16* Bb = WhT + (size_t)b * F_ * N_;
    v8f acc[4] = {v8f{}, v8f{}, v8f{}, v8f{}};

    for (int j0 = 0; j0 < N_; j0 += 32) {
        // ---- A fragment: 16x32 bf16, lane holds K = 8*hi+{0..7} and 16+8*hi+{0..7}
        int a0 = j0 + 8 * hi;
        int a1 = a0 + 16;
        float ysA[8], ysB[8], rsA[8], rsB[8];
        LD8(ysA, yl + a0);  LD8(ysB, yl + a1);
        LD8(rsA, rl + a0);  LD8(rsB, rl + a1);
        v16bf af = {};
#pragma unroll
        for (int q = 0; q < 8; ++q) {
            af[q]     = (__bf16)(att_val(xv + ysA[q]) * rsA[q]);
            af[q + 8] = (__bf16)(att_val(xv + ysB[q]) * rsB[q]);
        }
        // ---- B fragments: 32x16 bf16 per n-tile, lane holds K = 8*hi+{0..7} and 16+8*hi+{0..7}, col = mloc
        int bb = j0 + 8 * hi;
#pragma unroll
        for (int t = 0; t < 4; ++t) {
            const __bf16* brow = Bb + (size_t)(16 * t + mloc) * N_ + bb;
            v8bf lo = *(const v8bf*)brow;
            v8bf hv = *(const v8bf*)(brow + 16);
            v16bf bf = {};
#pragma unroll
            for (int q = 0; q < 8; ++q) { bf[q] = lo[q]; bf[q + 8] = hv[q]; }
            acc[t] = __builtin_amdgcn_wmma_f32_16x16x32_bf16(
                false, af, false, bf, (short)0, acc[t], false, false);
        }
    }

    // ---- Epilogue: ELU, scatter to out[b][i0+m][16t+n]
    float* ob = out + ((size_t)b * N_ + i0) * F_;
#pragma unroll
    for (int t = 0; t < 4; ++t) {
#pragma unroll
        for (int r = 0; r < 8; ++r) {
            float v = acc[t][r];
            float res = (v > 0.f) ? v : (__builtin_amdgcn_exp2f(v * LOG2E) - 1.0f);
            ob[(size_t)(r + 8 * hi) * F_ + 16 * t + mloc] = res;
        }
    }
}

extern "C" void kernel_launch(void* const* d_in, const int* in_sizes, int n_in,
                              void* d_out, int out_size, void* d_ws, size_t ws_size,
                              hipStream_t stream) {
    const float* h = (const float*)d_in[0];   // [16,2048,64]
    const float* W = (const float*)d_in[1];   // [64,64]
    const float* a = (const float*)d_in[2];   // [128,1]
    float* out = (float*)d_out;               // [16,2048,64] f32

    char*  ws  = (char*)d_ws;
    __bf16* WhT = (__bf16*)ws;                                 // [B][F][N] bf16, 4 MiB
    float* x   = (float*)(ws + (size_t)B_ * F_ * N_ * sizeof(__bf16));
    float* y   = x + (size_t)B_ * N_;
    float* rD  = y + (size_t)B_ * N_;

    k1_project <<<dim3((B_ * N_) / 128), dim3(128), 0, stream>>>(h, W, a, WhT, x, y);
    k2_colsum  <<<dim3(N_ / 256, B_),    dim3(256), 0, stream>>>(x, y, rD);
    k3_attn_gemm<<<dim3(N_ / 64, B_),    dim3(128), 0, stream>>>(WhT, x, y, rD, out);
}
